// DMoN_11562051960853
// MI455X (gfx1250) — compile-verified, matrix operand order checked by reading, not run on hardware
//
#include <hip/hip_runtime.h>
#include <math.h>

// ---------------------------------------------------------------------------
// DMoN pooling forward for MI455X (gfx1250, wave32).
//
// Key algebraic reductions (avoid materializing AS = A@S):
//   trace(graph_pooled) = sum_e w_e * dot(S[row_e], S[col_e])
//   trace(normalizer)   = ||S.T @ deg||^2 / (2*n_edges)
//
// Kernels:
//   k_init    : zero workspace
//   k_deg     : deg[c] += w_e (atomics; 400KB array is L2-resident), n_edges
//   k_assign  : S = softmax(F@W.T + b) via v_wmma_f32_16x16x32_f16,
//               fused cluster_sizes and wd = S.T@deg column reductions
//   k_edge    : edge trace, one wave per edge (S gathers L2-resident)
//   k_pool    : P = S.T @ F via WMMA; chunks DMA'd to LDS by the Tensor Data
//               Mover (TENSOR_LOAD_TO_LDS, double-buffered, s_wait_tensorcnt)
//   k_final   : losses + selu(P / cs)
// ---------------------------------------------------------------------------

#define N_NODES 100000
#define N_EDGES 3200000
#define D_FEAT  128
#define K_CL    64

typedef __attribute__((ext_vector_type(16))) _Float16 v16h;
typedef __attribute__((ext_vector_type(8)))  float    v8f;

// workspace layout (float offsets)
#define WS_DEG   0
#define WS_CS    100000
#define WS_WD    100064
#define WS_NE    100128
#define WS_TR    100129
#define WS_POOL  100160
#define WS_TOTAL (100160 + 64*128)

// output layout (float offsets): features_pooled, assignments, spectral, collapse
#define OUT_FP       0
#define OUT_ASSIGN   (64*128)
#define OUT_SPECTRAL (64*128 + N_NODES*K_CL)
#define OUT_COLLAPSE (OUT_SPECTRAL + 1)

#if defined(__has_builtin)
#if __has_builtin(__builtin_amdgcn_tensor_load_to_lds) && \
    __has_builtin(__builtin_amdgcn_s_wait_tensorcnt)
#define HAVE_TDM 1
#endif
#endif

__global__ __launch_bounds__(256) void k_init(float* __restrict__ ws) {
    int i = blockIdx.x * blockDim.x + threadIdx.x;
    if (i < WS_TOTAL) ws[i] = 0.0f;
}

__global__ __launch_bounds__(256) void k_deg(const float* __restrict__ ev,
                                             const int*   __restrict__ ec,
                                             float* __restrict__ deg,
                                             float* __restrict__ ne) {
    __shared__ float red[256];
    float s = 0.0f;
    for (int i = blockIdx.x * blockDim.x + threadIdx.x; i < N_EDGES;
         i += gridDim.x * blockDim.x) {
        float v = ev[i];
        atomicAdd(deg + ec[i], v);
        s += v;
    }
    red[threadIdx.x] = s;
    __syncthreads();
    for (int w = 128; w > 0; w >>= 1) {
        if (threadIdx.x < w) red[threadIdx.x] += red[threadIdx.x + w];
        __syncthreads();
    }
    if (threadIdx.x == 0) atomicAdd(ne, red[0]);
}

// S = softmax(F @ W.T + b); fused cs[j] = sum_n S[n,j], wd[j] = sum_n deg[n]*S[n,j].
__global__ __launch_bounds__(256) void k_assign(const float* __restrict__ F,
                                                const float* __restrict__ W,
                                                const float* __restrict__ bias,
                                                const float* __restrict__ deg,
                                                float* __restrict__ assign_out,
                                                float* __restrict__ cs_g,
                                                float* __restrict__ wd_g) {
    __shared__ float lds[8][16][65];   // padded rows -> conflict-free row softmax
    __shared__ float degs_sh[128];

    const int lane = threadIdx.x & 31;
    const int wv   = threadIdx.x >> 5;
    const int m16  = lane & 15;
    const int h    = lane >> 4;
    const int tile = blockIdx.x * 8 + wv;
    const bool valid = (tile * 16) < N_NODES;
    const int node_base = tile * 16;

    if (valid) {
        v8f acc[4];
        #pragma unroll
        for (int jt = 0; jt < 4; ++jt) acc[jt] = (v8f)(0.0f);

        const int row = node_base + m16;          // A-matrix row (lane M = lane%16)
        #pragma unroll
        for (int kc = 0; kc < 4; ++kc) {          // K = 128 in chunks of 32
            v16h a;
            #pragma unroll
            for (int v = 0; v < 8; ++v) {         // ISA 16-bit A layout, §7.12.2
                int k0 = kc * 32 + 16 * (v >> 2) + 8 * h + 2 * (v & 3);
                float2 f2 = *(const float2*)(F + (size_t)row * D_FEAT + k0);
                a[2 * v]     = (_Float16)f2.x;
                a[2 * v + 1] = (_Float16)f2.y;
            }
            #pragma unroll
            for (int jt = 0; jt < 4; ++jt) {      // 64 clusters in 4 N-tiles
                v16h bm;                          // B[kk][j] = W[j][kk]
                const int col = jt * 16 + m16;
                #pragma unroll
                for (int v = 0; v < 8; ++v) {
                    int k0 = kc * 32 + 16 * (v >> 2) + 8 * h + 2 * (v & 3);
                    float2 w2 = *(const float2*)(W + (size_t)col * D_FEAT + k0);
                    bm[2 * v]     = (_Float16)w2.x;
                    bm[2 * v + 1] = (_Float16)w2.y;
                }
                acc[jt] = __builtin_amdgcn_wmma_f32_16x16x32_f16(
                    false, a, false, bm, (short)0, acc[jt], false, false);
            }
        }
        #pragma unroll
        for (int jt = 0; jt < 4; ++jt)
            #pragma unroll
            for (int v = 0; v < 8; ++v)
                lds[wv][v + 8 * h][jt * 16 + m16] = acc[jt][v] + bias[jt * 16 + m16];
    } else {
        for (int i = lane; i < 16 * 64; i += 32) lds[wv][i >> 6][i & 63] = 0.0f;
    }
    if (h == 0) degs_sh[wv * 16 + m16] = valid ? deg[node_base + m16] : 0.0f;
    __syncthreads();

    if (valid && h == 0) {                        // row softmax (lane owns a row)
        float mx = -1e30f;
        #pragma unroll
        for (int j = 0; j < 64; ++j) mx = fmaxf(mx, lds[wv][m16][j]);
        float s = 0.0f;
        #pragma unroll
        for (int j = 0; j < 64; ++j) {
            float e = __expf(lds[wv][m16][j] - mx);
            lds[wv][m16][j] = e;
            s += e;
        }
        float inv = 1.0f / s;
        #pragma unroll
        for (int j = 0; j < 64; ++j) lds[wv][m16][j] *= inv;
    }
    __syncthreads();

    if (valid) {                                  // coalesced tile write
        float* dst = assign_out + (size_t)node_base * K_CL;
        for (int i = lane; i < 16 * 64; i += 32)
            dst[i] = lds[wv][i >> 6][i & 63];
    }
    if (threadIdx.x < 64) {                       // block column reduction
        const int j = threadIdx.x;
        float cs = 0.0f, wd = 0.0f;
        #pragma unroll 4
        for (int r = 0; r < 128; ++r) {
            float p = lds[r >> 4][r & 15][j];
            cs += p;
            wd += degs_sh[r] * p;
        }
        atomicAdd(cs_g + j, cs);
        atomicAdd(wd_g + j, wd);
    }
}

__global__ __launch_bounds__(256) void k_edge(const float* __restrict__ S,
                                              const float* __restrict__ ev,
                                              const int*   __restrict__ er,
                                              const int*   __restrict__ ec,
                                              float* __restrict__ tr_g) {
    const int lane = threadIdx.x & 31;
    const int wvg  = (blockIdx.x * blockDim.x + threadIdx.x) >> 5;
    const int nwv  = (gridDim.x * blockDim.x) >> 5;
    float local = 0.0f;
    for (int e = wvg; e < N_EDGES; e += nwv) {
        const int r = er[e], c = ec[e];
        const float w = ev[e];
        const float* sr = S + (size_t)r * K_CL;
        const float* sc = S + (size_t)c * K_CL;
        float t = sr[lane] * sc[lane] + sr[lane + 32] * sc[lane + 32];
        #pragma unroll
        for (int off = 16; off > 0; off >>= 1) t += __shfl_xor(t, off, 32);
        local += w * t;
    }
    if (lane == 0) atomicAdd(tr_g, local);
}

#if HAVE_TDM
typedef __attribute__((ext_vector_type(4))) unsigned int u32x4;
typedef __attribute__((ext_vector_type(8))) int          i32x8;
typedef __attribute__((ext_vector_type(4))) int          i32x4;

// Issue one 2-D TENSOR_LOAD_TO_LDS: tile of `rows` x `row_elems` f32 starting
// at gsrc (row-major, row stride == row_elems) into LDS at lds_dst.
// D# packing per CDNA5 ISA §8.3 (group0) / §8.4 (group1); groups 2/3 unused (2-D).
// This toolchain's builtin is the 6-arg form:
//   (u32x4 g0, i32x8 g1, i32x4 g2, i32x4 g3, i32x8, i32 cpol)
__device__ __forceinline__ void tdm_load_2d(const float* gsrc, void* lds_dst,
                                            unsigned row_elems, unsigned rows,
                                            unsigned tensor_rows) {
    unsigned long long ga = (unsigned long long)(uintptr_t)gsrc;
    unsigned lds = (unsigned)(uintptr_t)lds_dst;       // low 32 bits = LDS offset
    u32x4 g0;
    g0.x = 1u;                                          // count=1, no gather
    g0.y = lds;                                         // lds_addr
    g0.z = (unsigned)ga;                                // global_addr[31:0]
    g0.w = (unsigned)((ga >> 32) & 0x01FFFFFFull)       // global_addr[56:32]
           | 0x80000000u;                               // type=2 ("image")
    i32x8 g1;
    g1[0] = 0x00020000;                                 // data_size=2 -> 4 bytes
    g1[1] = (int)((row_elems & 0xFFFFu) << 16);         // tensor_dim0[15:0]
    g1[2] = (int)((row_elems >> 16) |
                  ((tensor_rows & 0xFFFFu) << 16));     // dim0[31:16] | dim1[15:0]
    g1[3] = (int)((tensor_rows >> 16) |
                  (row_elems << 16));                   // dim1[31:16] | tile_dim0
    g1[4] = (int)rows;                                  // tile_dim1 (tile_dim2=0)
    g1[5] = (int)row_elems;                             // tensor_dim0_stride[31:0]
    g1[6] = 0;
    g1[7] = 0;
    i32x4 z4 = (i32x4)0;
    i32x8 z8 = (i32x8)0;
    __builtin_amdgcn_tensor_load_to_lds(g0, g1, z4, z4, z8, 0);
}
#endif

// P = S.T @ F ([64 x 128]); split-K over 32-node chunks. Chunks are loaded into
// double-buffered LDS by the TDM (wave 0 issues, TENSORcnt tracks, barrier
// publishes); f32->f16 conversion is folded into WMMA operand packing.
__global__ __launch_bounds__(256) void k_pool(const float* __restrict__ S,
                                              const float* __restrict__ F,
                                              float* __restrict__ pool) {
    __shared__ float Sf[2][32][64];    // 16 KB
    __shared__ float Ff[2][32][128];   // 32 KB

    const int lane = threadIdx.x & 31;
    const int wv   = threadIdx.x >> 5;
    const int m16  = lane & 15;
    const int h    = lane >> 4;

    v8f acc[4];
    #pragma unroll
    for (int ct = 0; ct < 4; ++ct) acc[ct] = (v8f)(0.0f);

    const int nTiles = N_NODES / 32;   // 3125
    int cur = 0;

#if HAVE_TDM
    if (wv == 0 && (int)blockIdx.x < nTiles) {   // prologue: chunk 0 -> buf 0
        const int nb0 = blockIdx.x * 32;
        tdm_load_2d(S + (size_t)nb0 * K_CL,   &Sf[0][0][0], K_CL,   32, N_NODES);
        tdm_load_2d(F + (size_t)nb0 * D_FEAT, &Ff[0][0][0], D_FEAT, 32, N_NODES);
    }
#endif

    for (int t = blockIdx.x; t < nTiles; t += gridDim.x) {
#if HAVE_TDM
        const int tn = t + (int)gridDim.x;
        if (wv == 0) {
            if (tn < nTiles) {                   // prefetch next chunk -> other buf
                const int nbn = tn * 32;
                tdm_load_2d(S + (size_t)nbn * K_CL,   &Sf[cur ^ 1][0][0], K_CL,   32, N_NODES);
                tdm_load_2d(F + (size_t)nbn * D_FEAT, &Ff[cur ^ 1][0][0], D_FEAT, 32, N_NODES);
                __builtin_amdgcn_s_wait_tensorcnt(2);  // current chunk's 2 loads done
            } else {
                __builtin_amdgcn_s_wait_tensorcnt(0);
            }
        }
        __syncthreads();                         // publish LDS to all 8 waves
#else
        const int nb = t * 32;
        for (int i = threadIdx.x; i < 32 * 64; i += 256)
            Sf[0][i >> 6][i & 63] = S[(size_t)(nb + (i >> 6)) * K_CL + (i & 63)];
        for (int i = threadIdx.x; i < 32 * 128; i += 256)
            Ff[0][i >> 7][i & 127] = F[(size_t)(nb + (i >> 7)) * D_FEAT + (i & 127)];
        __syncthreads();
#endif

        v16h bm;                                 // B[kk][n] = F[nb+kk][wv*16+n]
        #pragma unroll
        for (int v = 0; v < 8; ++v) {
            int k0 = 16 * (v >> 2) + 8 * h + 2 * (v & 3);
            bm[2 * v]     = (_Float16)Ff[cur][k0][wv * 16 + m16];
            bm[2 * v + 1] = (_Float16)Ff[cur][k0 + 1][wv * 16 + m16];
        }
        #pragma unroll
        for (int ct = 0; ct < 4; ++ct) {
            v16h a;                              // A[m][kk] = S[nb+kk][ct*16+m]
            #pragma unroll
            for (int v = 0; v < 8; ++v) {
                int k0 = 16 * (v >> 2) + 8 * h + 2 * (v & 3);
                a[2 * v]     = (_Float16)Sf[cur][k0][ct * 16 + m16];
                a[2 * v + 1] = (_Float16)Sf[cur][k0 + 1][ct * 16 + m16];
            }
            acc[ct] = __builtin_amdgcn_wmma_f32_16x16x32_f16(
                false, a, false, bm, (short)0, acc[ct], false, false);
        }
        __syncthreads();                         // done reading buf `cur`
#if HAVE_TDM
        cur ^= 1;
#endif
    }
    #pragma unroll
    for (int ct = 0; ct < 4; ++ct)
        #pragma unroll
        for (int v = 0; v < 8; ++v)
            atomicAdd(pool + (size_t)(ct * 16 + v + 8 * h) * D_FEAT + wv * 16 + m16,
                      acc[ct][v]);
}

__global__ __launch_bounds__(256) void k_final(const float* __restrict__ ws,
                                               float* __restrict__ out) {
    const float* cs   = ws + WS_CS;
    const float* wd   = ws + WS_WD;
    const float* pool = ws + WS_POOL;
    if (blockIdx.x == 0 && threadIdx.x == 0) {
        const float E2 = 2.0f * ws[WS_NE];
        float wd2 = 0.0f, cs2 = 0.0f;
        for (int j = 0; j < 64; ++j) { wd2 += wd[j] * wd[j]; cs2 += cs[j] * cs[j]; }
        const float tr_norm = wd2 / E2;
        out[OUT_SPECTRAL] = -(ws[WS_TR] - tr_norm) / E2;
        out[OUT_COLLAPSE] = 0.1f * (sqrtf(cs2) / (float)N_NODES * 8.0f - 1.0f);
    }
    const float scale = 1.0507009873554805f;
    const float alpha = 1.6732632423543772f;
    for (int i = blockIdx.x * blockDim.x + threadIdx.x; i < K_CL * D_FEAT;
         i += gridDim.x * blockDim.x) {
        const int k = i >> 7;
        float x = pool[i] / cs[k];
        out[OUT_FP + i] = (x > 0.0f) ? scale * x : scale * alpha * (__expf(x) - 1.0f);
    }
}

extern "C" void kernel_launch(void* const* d_in, const int* in_sizes, int n_in,
                              void* d_out, int out_size, void* d_ws, size_t ws_size,
                              hipStream_t stream) {
    const float* F    = (const float*)d_in[0];
    const float* W    = (const float*)d_in[1];
    const float* bias = (const float*)d_in[2];
    const float* ev   = (const float*)d_in[3];
    const int*   er   = (const int*)d_in[4];
    const int*   ec   = (const int*)d_in[5];
    float* out = (float*)d_out;
    float* ws  = (float*)d_ws;

    k_init<<<(WS_TOTAL + 255) / 256, 256, 0, stream>>>(ws);
    k_deg<<<2048, 256, 0, stream>>>(ev, ec, ws + WS_DEG, ws + WS_NE);
    k_assign<<<(N_NODES / 16 + 7) / 8, 256, 0, stream>>>(
        F, W, bias, ws + WS_DEG, out + OUT_ASSIGN, ws + WS_CS, ws + WS_WD);
    k_edge<<<1024, 256, 0, stream>>>(out + OUT_ASSIGN, ev, er, ec, ws + WS_TR);
    k_pool<<<625, 256, 0, stream>>>(out + OUT_ASSIGN, F, ws + WS_POOL);
    k_final<<<32, 256, 0, stream>>>(ws, out);
}
